// Classifier_GNN_23682449670459
// MI455X (gfx1250) — matrix operand order repaired and checked by reading, not compiled, not run on hardware
//
#include <hip/hip_runtime.h>
#include <hip/hip_bf16.h>
#include <stdint.h>

#define NN 50000
#define NE 800000
#define NTRAIN 40000

typedef __attribute__((ext_vector_type(16))) __bf16    v16bf;
typedef __attribute__((ext_vector_type(8)))  float     v8f;
typedef __attribute__((ext_vector_type(8)))  unsigned  v8u;

// ---------- device helpers ----------
__device__ __forceinline__ unsigned short f2bf(float f) {
  unsigned u = __float_as_uint(f);
  u += 0x7FFFu + ((u >> 16) & 1u);            // round-to-nearest-even
  return (unsigned short)(u >> 16);
}
__device__ __forceinline__ unsigned ord_enc(float f) {   // monotone float->uint
  unsigned u = __float_as_uint(f);
  return (u & 0x80000000u) ? ~u : (u | 0x80000000u);
}
__device__ __forceinline__ float ord_dec(unsigned o) {
  unsigned u = (o & 0x80000000u) ? (o ^ 0x80000000u) : ~o;
  return __uint_as_float(u);
}

// ---------- utility kernels ----------
__global__ void k_zero(unsigned* __restrict__ p, long n) {
  long i = (long)blockIdx.x * blockDim.x + threadIdx.x;
  if (i < n) p[i] = 0u;
}

// f32 [M,K] row-major -> bf16 [M,Kp] zero-padded
__global__ void k_cvt_x(const float* __restrict__ in, unsigned short* __restrict__ out,
                        int M, int K, int Kp) {
  long i = (long)blockIdx.x * blockDim.x + threadIdx.x;
  if (i >= (long)M * Kp) return;
  int m = (int)(i / Kp), k = (int)(i % Kp);
  out[i] = (k < K) ? f2bf(in[(long)m * K + k]) : (unsigned short)0;
}

// W f32 [K, Nl] -> Wt bf16 [Np, Kp]  (transposed, zero-padded)
__global__ void k_cvt_wt(const float* __restrict__ W, unsigned short* __restrict__ Wt,
                         int K, int Kp, int Nl, int Np) {
  long i = (long)blockIdx.x * blockDim.x + threadIdx.x;
  if (i >= (long)Np * Kp) return;
  int n = (int)(i / Kp), k = (int)(i % Kp);
  Wt[i] = (n < Nl && k < K) ? f2bf(W[(long)k * Nl + n]) : (unsigned short)0;
}

// ---------- bf16 WMMA GEMM: C[M,N] = A[M,K] * Bt[N,K]^T ----------
// One wave computes a 16 x (16*NT) strip: the A fragment is loaded once per
// K-step and reused across NT WMMAs (register-level A reuse).
// K multiple of 32; M multiple of 16; N multiple of 16*NT.
template <int NT>
__global__ void __launch_bounds__(256)
k_gemm_bf16(const unsigned short* __restrict__ A,
            const unsigned short* __restrict__ Bt,
            float* __restrict__ C, int M, int N, int K) {
  int wid  = (blockIdx.x * blockDim.x + threadIdx.x) >> 5;
  int lane = threadIdx.x & 31;
  int tn_cnt = N / (16 * NT);
  int tm = wid / tn_cnt, tn0 = (wid % tn_cnt) * NT;
  if (tm * 16 >= M) return;
  int g  = lane >> 4;        // half-wave group
  int mr = lane & 15;        // A row / B column within tile
  const unsigned* Ap = (const unsigned*)(A + (size_t)(tm * 16 + mr) * K); // bf16 pairs
  const unsigned* Bp[NT];
#pragma unroll
  for (int t = 0; t < NT; ++t)
    Bp[t] = (const unsigned*)(Bt + (size_t)((tn0 + t) * 16 + mr) * K);

  v8f acc[NT];
#pragma unroll
  for (int t = 0; t < NT; ++t)
    acc[t] = (v8f){0.f, 0.f, 0.f, 0.f, 0.f, 0.f, 0.f, 0.f};

  for (int k0 = 0; k0 < K; k0 += 32) {
    int pb = k0 >> 1;                    // base in bf16-pair units
    // 16-bit A 16x32 layout: v0..3 -> K = 8g+2v{+1}; v4..7 -> K = 16+8g+2(v-4){+1}
    v8u au;
#pragma unroll
    for (int v = 0; v < 4; ++v) {
      au[v]     = Ap[pb + 4 * g + v];
      au[v + 4] = Ap[pb + 8 + 4 * g + v];
    }
    v16bf av = __builtin_bit_cast(v16bf, au);
    // 16-bit B 32x16 layout (same K order as A): v0..3 -> K = 8g+2v{+1}; v4..7 -> K = 16+8g+2(v-4){+1}
#pragma unroll
    for (int t = 0; t < NT; ++t) {
      v8u bu;
#pragma unroll
      for (int v = 0; v < 8; ++v)
        bu[v] = Bp[t][pb + 4 * g + v + (v & 4)];
      v16bf bv = __builtin_bit_cast(v16bf, bu);
      acc[t] = __builtin_amdgcn_wmma_f32_16x16x32_bf16(
          /*neg_a=*/false, av, /*neg_b=*/false, bv,
          /*c_mod=*/(short)0, acc[t], /*reuse_a=*/false, /*reuse_b=*/false);
    }
  }
  // C/D layout: lane L, VGPR r -> C[8g + r][L&15]
#pragma unroll
  for (int t = 0; t < NT; ++t) {
    size_t base = (size_t)(tm * 16 + 8 * g) * N + (size_t)(tn0 + t) * 16 + mr;
#pragma unroll
    for (int r = 0; r < 8; ++r)
      C[base + (size_t)r * N] = acc[t][r];
  }
}

// ---------- edge kernels ----------
// one wave per (edge, head): leaky_relu(xl[src]+xr[dst]) . att  -> logit; atomic segment max
__global__ void k_edge_logits(const float* __restrict__ xl, const float* __restrict__ xr,
                              const int* __restrict__ src, const int* __restrict__ dst,
                              const float* __restrict__ att,
                              float* __restrict__ logits, unsigned* __restrict__ nmax,
                              int E, int H, int C, int ld) {
  long idx = (long)blockIdx.x * blockDim.x + threadIdx.x;
  long wid = idx >> 5;
  int lane = (int)(idx & 31);
  if (wid >= (long)E * H) return;
  int e = (int)(wid / H), h = (int)(wid % H);
  int s = src[e], d = dst[e];
  const float* pl = xl + (size_t)s * ld + h * C;
  const float* pr = xr + (size_t)d * ld + h * C;
  const float* pa = att + h * C;
  float sum = 0.f;
  for (int c = lane; c < C; c += 32) {
    float v = pl[c] + pr[c];
    v = (v > 0.f) ? v : 0.2f * v;
    sum += v * pa[c];
  }
#pragma unroll
  for (int off = 16; off > 0; off >>= 1) sum += __shfl_xor(sum, off, 32);
  if (lane == 0) {
    logits[(size_t)e * H + h] = sum;
    atomicMax(&nmax[(size_t)d * H + h], ord_enc(sum));
  }
}

// one thread per (edge, head): a = exp(logit - max[dst]); denom += a; deg += 1 (h==0)
__global__ void k_edge_exp(const float* __restrict__ logits, const int* __restrict__ dst,
                           const unsigned* __restrict__ nmax,
                           float* __restrict__ expa, float* __restrict__ denom,
                           float* __restrict__ deg, int E, int H) {
  long i = (long)blockIdx.x * blockDim.x + threadIdx.x;
  if (i >= (long)E * H) return;
  int e = (int)(i / H), h = (int)(i % H);
  int d = dst[e];
  float m = ord_dec(nmax[(size_t)d * H + h]);
  float a = expf(logits[i] - m);
  expa[i] = a;
  atomicAdd(&denom[(size_t)d * H + h], a);
  if (h == 0) atomicAdd(&deg[d], 1.0f);
}

// one wave per (edge, head): out[dst] += xl[src] * alpha
__global__ void k_edge_msg(const float* __restrict__ xl, const float* __restrict__ expa,
                           const float* __restrict__ denom,
                           const int* __restrict__ src, const int* __restrict__ dst,
                           float* __restrict__ acc, int E, int H, int C, int ld, int ldo) {
  long idx = (long)blockIdx.x * blockDim.x + threadIdx.x;
  long wid = idx >> 5;
  int lane = (int)(idx & 31);
  if (wid >= (long)E * H) return;
  int e = (int)(wid / H), h = (int)(wid % H);
  int s = src[e], d = dst[e];
  float alpha = expa[(size_t)e * H + h] / denom[(size_t)d * H + h];
  const float* pl = xl + (size_t)s * ld + h * C;
  float* po = acc + (size_t)d * ldo + h * C;
  for (int c = lane; c < C; c += 32)
    atomicAdd(&po[c], pl[c] * alpha);
}

// out = out / max(deg,1) + bias
__global__ void k_finalize(float* __restrict__ acc, const float* __restrict__ deg,
                           const float* __restrict__ bias, int M, int HC) {
  long i = (long)blockIdx.x * blockDim.x + threadIdx.x;
  if (i >= (long)M * HC) return;
  int m = (int)(i / HC), j = (int)(i % HC);
  float dg = deg[m];
  dg = (dg > 1.f) ? dg : 1.f;
  acc[i] = acc[i] / dg + bias[j];
}

// d_out[0:n) = sigmoid(h3[train_idx]);  d_out[n:2n) = y[train_idx]
__global__ void k_gather(const float* __restrict__ h3, const float* __restrict__ y,
                         const int* __restrict__ tidx, float* __restrict__ out, int n) {
  int i = blockIdx.x * blockDim.x + threadIdx.x;
  if (i >= n) return;
  int t = tidx[i];
  float v = h3[t];
  out[i]     = 1.f / (1.f + expf(-v));
  out[n + i] = y[t];
}

// ---------- host orchestration ----------
static inline void* wsa(char*& base, size_t bytes) {
  void* p = (void*)base;
  base += (bytes + 255) & ~(size_t)255;
  return p;
}
static inline unsigned gdiv(long n, long b) { return (unsigned)((n + b - 1) / b); }

static void launch_gemm(const unsigned short* A, const unsigned short* Bt, float* C,
                        int M, int N, int K, hipStream_t s) {
  if (N % 64 == 0) {
    long waves = (long)(M / 16) * (N / 64);
    k_gemm_bf16<4><<<gdiv(waves * 32, 256), 256, 0, s>>>(A, Bt, C, M, N, K);
  } else {
    long waves = (long)(M / 16) * (N / 16);
    k_gemm_bf16<1><<<gdiv(waves * 32, 256), 256, 0, s>>>(A, Bt, C, M, N, K);
  }
}

static void run_layer(const float* hin, int M, int Fin, int Kp,
                      const float* Wl, const float* Wr,
                      int H, int C, int HCp,
                      const float* att, const float* bias,
                      const int* src, const int* dst, int E,
                      unsigned short* xbf, unsigned short* wlt, unsigned short* wrt,
                      float* xl, float* xr, float* logits, float* expa,
                      unsigned* nmax, float* denom, float* deg,
                      float* out, hipStream_t s) {
  int HC = H * C;
  long nx = (long)M * Kp;
  k_cvt_x<<<gdiv(nx, 256), 256, 0, s>>>(hin, xbf, M, Fin, Kp);
  long nw = (long)HCp * Kp;
  k_cvt_wt<<<gdiv(nw, 256), 256, 0, s>>>(Wl, wlt, Fin, Kp, HC, HCp);
  k_cvt_wt<<<gdiv(nw, 256), 256, 0, s>>>(Wr, wrt, Fin, Kp, HC, HCp);
  launch_gemm(xbf, wlt, xl, M, HCp, Kp, s);
  launch_gemm(xbf, wrt, xr, M, HCp, Kp, s);
  long nz = (long)M * H;
  k_zero<<<gdiv(nz, 256), 256, 0, s>>>(nmax, nz);
  k_zero<<<gdiv(nz, 256), 256, 0, s>>>((unsigned*)denom, nz);
  k_zero<<<gdiv(M, 256), 256, 0, s>>>((unsigned*)deg, M);
  long no = (long)M * HC;
  k_zero<<<gdiv(no, 256), 256, 0, s>>>((unsigned*)out, no);
  long ew = (long)E * H;
  k_edge_logits<<<gdiv(ew * 32, 256), 256, 0, s>>>(xl, xr, src, dst, att, logits, nmax, E, H, C, HCp);
  k_edge_exp<<<gdiv(ew, 256), 256, 0, s>>>(logits, dst, nmax, expa, denom, deg, E, H);
  k_edge_msg<<<gdiv(ew * 32, 256), 256, 0, s>>>(xl, expa, denom, src, dst, out, E, H, C, HCp, HC);
  k_finalize<<<gdiv(no, 256), 256, 0, s>>>(out, deg, bias, M, HC);
}

extern "C" void kernel_launch(void* const* d_in, const int* in_sizes, int n_in,
                              void* d_out, int out_size, void* d_ws, size_t ws_size,
                              hipStream_t stream) {
  const float* x    = (const float*)d_in[0];
  const int*   edge = (const int*)d_in[1];
  const float* y    = (const float*)d_in[2];
  const int*   tidx = (const int*)d_in[3];
  const float* W1l  = (const float*)d_in[4];
  const float* W1r  = (const float*)d_in[5];
  const float* a1   = (const float*)d_in[6];
  const float* b1   = (const float*)d_in[7];
  const float* W2l  = (const float*)d_in[8];
  const float* W2r  = (const float*)d_in[9];
  const float* a2   = (const float*)d_in[10];
  const float* b2   = (const float*)d_in[11];
  const float* W3l  = (const float*)d_in[12];
  const float* W3r  = (const float*)d_in[13];
  const float* a3   = (const float*)d_in[14];
  const float* b3   = (const float*)d_in[15];
  const int* src = edge;        // edge_index row 0
  const int* dst = edge + NE;   // edge_index row 1

  char* w = (char*)d_ws;
  unsigned short* xbf = (unsigned short*)wsa(w, (size_t)NN * 256 * 2);
  unsigned short* wlt = (unsigned short*)wsa(w, (size_t)256 * 256 * 2);
  unsigned short* wrt = (unsigned short*)wsa(w, (size_t)256 * 256 * 2);
  float* xl     = (float*)wsa(w, (size_t)NN * 256 * 4);
  float* xr     = (float*)wsa(w, (size_t)NN * 256 * 4);
  float* hA     = (float*)wsa(w, (size_t)NN * 256 * 4);
  float* hB     = (float*)wsa(w, (size_t)NN * 128 * 4);
  float* logits = (float*)wsa(w, (size_t)NE * 2 * 4);
  float* expa   = (float*)wsa(w, (size_t)NE * 2 * 4);
  unsigned* nmax = (unsigned*)wsa(w, (size_t)NN * 2 * 4);
  float* denom  = (float*)wsa(w, (size_t)NN * 2 * 4);
  float* deg    = (float*)wsa(w, (size_t)NN * 4);

  // Layer 1: 129 -> (H=2, C=128), K padded 129->160
  run_layer(x, NN, 129, 160, W1l, W1r, 2, 128, 256, a1, b1, src, dst, NE,
            xbf, wlt, wrt, xl, xr, logits, expa, nmax, denom, deg, hA, stream);
  // Layer 2: 256 -> (H=1, C=128)
  run_layer(hA, NN, 256, 256, W2l, W2r, 1, 128, 128, a2, b2, src, dst, NE,
            xbf, wlt, wrt, xl, xr, logits, expa, nmax, denom, deg, hB, stream);
  // Layer 3: 128 -> (H=1, C=1); GEMM N padded to 16
  run_layer(hB, NN, 128, 128, W3l, W3r, 1, 1, 16, a3, b3, src, dst, NE,
            xbf, wlt, wrt, xl, xr, logits, expa, nmax, denom, deg, hA, stream);

  k_gather<<<gdiv(NTRAIN, 256), 256, 0, stream>>>(hA, y, tidx, (float*)d_out, NTRAIN);
}
